// EnhancedSympNet_55808805045037
// MI455X (gfx1250) — matrix operand order repaired and checked by reading, not compiled, not run on hardware
//
#include <hip/hip_runtime.h>
#include <math.h>

typedef _Float16 half_t;
typedef _Float16 h8   __attribute__((ext_vector_type(8)));
typedef _Float16 v16h __attribute__((ext_vector_type(16)));
typedef float    v8f  __attribute__((ext_vector_type(8)));

// Global-address-space pointer types: force global_load_b128 (LOADcnt only,
// saddr form) instead of flat_load (which also ties up DScnt / LDS pipe).
typedef const __attribute__((address_space(1))) _Float16* gcptr;
typedef const __attribute__((address_space(1))) h8*       gch8p;

#define HDIM   256
#define NSTEPS 32
#define TILEB  32
#define HSTR   264      // LDS row stride in halfs: 528B -> rotates 4 banks/row, conflict-free b128 reads
#define BETA_C 0.1f

// Launder a uniform pointer through an opaque asm (kept in an SGPR pair) so the
// compiler cannot hoist the loop-invariant weight-fragment loads out of the
// time loop and spill them to scratch; returns a global-AS pointer.
__device__ __forceinline__ gcptr launder_g(const half_t* p) {
  gcptr q = (gcptr)p;
  asm volatile("" : "+s"(q));
  return q;
}

// ---- WMMA fragment loaders (CDNA5 16x16x32 f16 layouts) ----
// A (16x32, MxK): lane m=lane&15, g=lane>>4; half h: K = (h>>3)*16 + 8g + (h&7)
__device__ __forceinline__ v16h load_A_frag(const half_t* __restrict__ row, int kBase, int g) {
  h8 lo = *(const h8*)(row + kBase + 8 * g);
  h8 hi = *(const h8*)(row + kBase + 8 * g + 16);
  v16h r;
#pragma unroll
  for (int i = 0; i < 8; ++i) { r[i] = lo[i]; r[8 + i] = hi[i]; }
  return r;
}
// B (32x16, KxN): lane n=lane&15, g=lane>>4; half h: K = (h>>3)*16 + 8g + (h&7)
__device__ __forceinline__ v16h load_B_fragG(gcptr row, int kBase, int g) {
  gch8p p = (gch8p)(row + kBase + 8 * g);
  h8 lo = p[0];
  h8 hi = p[2];
  v16h r;
#pragma unroll
  for (int i = 0; i < 8; ++i) { r[i] = lo[i]; r[8 + i] = hi[i]; }
  return r;
}

// One wave computes a 2x2 block of 16x16 tiles:
//   rows [0,16) and [16,32) of the LDS A tile, columns [nB,nB+16) and [nB+16,nB+32).
// A: LDS, 32 rows, stride HSTR. B: global f16, rows indexed by N, stride HDIM.
__device__ __forceinline__ void gemm4(const half_t* __restrict__ A,
                                      gcptr B,
                                      const float* __restrict__ bias,
                                      int lane, int nB,
                                      v8f& c00, v8f& c01, v8f& c10, v8f& c11)
{
  const int nl = lane & 15;
  const int g  = lane >> 4;
  const float bv0 = bias ? bias[nB + nl]      : 0.0f;
  const float bv1 = bias ? bias[nB + 16 + nl] : 0.0f;
#pragma unroll
  for (int v = 0; v < 8; ++v) { c00[v] = bv0; c01[v] = bv1; c10[v] = bv0; c11[v] = bv1; }
  const half_t* arow0 = A + nl * HSTR;
  const half_t* arow1 = A + (16 + nl) * HSTR;
  gcptr brow0 = B + (size_t)(nB + nl) * HDIM;
  gcptr brow1 = B + (size_t)(nB + 16 + nl) * HDIM;
#pragma unroll
  for (int kk = 0; kk < 8; ++kk) {
    const int kBase = kk * 32;
    v16h a0 = load_A_frag(arow0, kBase, g);
    v16h a1 = load_A_frag(arow1, kBase, g);
    v16h b0 = load_B_fragG(brow0, kBase, g);
    v16h b1 = load_B_fragG(brow1, kBase, g);
    c00 = __builtin_amdgcn_wmma_f32_16x16x32_f16(false, a0, false, b0, (short)0, c00, false, false);
    c01 = __builtin_amdgcn_wmma_f32_16x16x32_f16(false, a0, false, b1, (short)0, c01, false, false);
    c10 = __builtin_amdgcn_wmma_f32_16x16x32_f16(false, a1, false, b0, (short)0, c10, false, false);
    c11 = __builtin_amdgcn_wmma_f32_16x16x32_f16(false, a1, false, b1, (short)0, c11, false, false);
  }
}

// Convert W2/W3 to f16, row-major and transposed, into workspace.
__global__ __launch_bounds__(256) void prep_weights(
    const float* __restrict__ W2, const float* __restrict__ W3,
    half_t* __restrict__ w2h, half_t* __restrict__ w2t,
    half_t* __restrict__ w3h, half_t* __restrict__ w3t)
{
  const int r = blockIdx.x & 255;
  const bool second = blockIdx.x >= 256;
  const float* W  = second ? W3 : W2;
  half_t* wh = second ? w3h : w2h;
  half_t* wt = second ? w3t : w2t;
  const int c = threadIdx.x;
  const float v = W[r * HDIM + c];
  wh[r * HDIM + c] = (half_t)v;
  wt[c * HDIM + r] = (half_t)v;
}

__global__ __launch_bounds__(256) void sympnet_kernel(
    const float* __restrict__ t_eval, const float* __restrict__ state0,
    const float* __restrict__ W1, const float* __restrict__ b1,
    const float* __restrict__ b2, const float* __restrict__ b3,
    const float* __restrict__ W4, const float* __restrict__ scale_p,
    const half_t* __restrict__ w2h, const half_t* __restrict__ w2t,
    const half_t* __restrict__ w3h, const half_t* __restrict__ w3t,
    float* __restrict__ out)
{
  __shared__ __attribute__((aligned(16))) half_t sH1[TILEB * HSTR];
  __shared__ __attribute__((aligned(16))) half_t sH2[TILEB * HSTR];
  __shared__ __attribute__((aligned(16))) half_t sDZ[TILEB * HSTR];
  __shared__ float sS[TILEB][4];
  __shared__ float sG[TILEB][4];

  const int tid  = threadIdx.x;
  const int lane = tid & 31;
  const int wv   = tid >> 5;          // 0..7, owns columns [32wv, 32wv+32)
  const int nl   = lane & 15;
  const int g    = lane >> 4;
  const int nB   = wv * 32;
  const int b0   = blockIdx.x * TILEB;

  const float dt  = t_eval[1] - t_eval[0];
  const float scl = scale_p[0];

  if (tid < TILEB * 4) {
    const int row = tid >> 2, c = tid & 3;
    const float v = state0[(size_t)(b0 + row) * 4 + c];
    sS[row][c] = v;
    out[(size_t)(b0 + row) * (NSTEPS * 4) + c] = v;   // t = 0
  }
  __syncthreads();

  for (int t = 1; t < NSTEPS; ++t) {
    // ---- layer 1 forward (K = 4, plain VALU): 8 threads/row x 32 cols ----
    {
      const int row = tid >> 3;            // 0..31
      const int cb  = (tid & 7) * 32;
      const float s0 = sS[row][0], s1 = sS[row][1], s2 = sS[row][2], s3 = sS[row][3];
#pragma unroll
      for (int j = 0; j < 32; ++j) {
        const int n = cb + j;
        const float* wr = W1 + n * 4;
        const float z = b1[n] + s0 * wr[0] + s1 * wr[1] + s2 * wr[2] + s3 * wr[3];
        sH1[row * HSTR + n] = (half_t)tanhf(z);
      }
    }
    __syncthreads();

    v8f c00, c01, c10, c11;
    // ---- layer 2 forward: h2 = tanh(h1 @ W2^T + b2) ----
    gemm4(sH1, launder_g(w2h), b2, lane, nB, c00, c01, c10, c11);
#pragma unroll
    for (int v = 0; v < 8; ++v) {
      const int M0 = v + 8 * g;
      const int M1 = 16 + M0;
      sH2[M0 * HSTR + nB + nl]      = (half_t)tanhf(c00[v]);
      sH2[M0 * HSTR + nB + 16 + nl] = (half_t)tanhf(c01[v]);
      sH2[M1 * HSTR + nB + nl]      = (half_t)tanhf(c10[v]);
      sH2[M1 * HSTR + nB + 16 + nl] = (half_t)tanhf(c11[v]);
    }
    __syncthreads();

    // ---- layer 3 forward + dz3 = W4 * (1 - h3^2) ----
    gemm4(sH2, launder_g(w3h), b3, lane, nB, c00, c01, c10, c11);
    {
      const float w40 = W4[nB + nl];
      const float w41 = W4[nB + 16 + nl];
#pragma unroll
      for (int v = 0; v < 8; ++v) {
        const int M0 = v + 8 * g;
        const int M1 = 16 + M0;
        float t0 = tanhf(c00[v]);
        float t1 = tanhf(c01[v]);
        float t2 = tanhf(c10[v]);
        float t3 = tanhf(c11[v]);
        sDZ[M0 * HSTR + nB + nl]      = (half_t)((1.0f - t0 * t0) * w40);
        sDZ[M0 * HSTR + nB + 16 + nl] = (half_t)((1.0f - t1 * t1) * w41);
        sDZ[M1 * HSTR + nB + nl]      = (half_t)((1.0f - t2 * t2) * w40);
        sDZ[M1 * HSTR + nB + 16 + nl] = (half_t)((1.0f - t3 * t3) * w41);
      }
    }
    __syncthreads();

    // ---- backward: dh2 = dz3 @ W3 ; dz2 = dh2 * (1 - h2^2) ----
    gemm4(sDZ, launder_g(w3t), nullptr, lane, nB, c00, c01, c10, c11);
    __syncthreads();   // all waves finished reading sDZ
#pragma unroll
    for (int v = 0; v < 8; ++v) {
      const int M0 = v + 8 * g;
      const int M1 = 16 + M0;
      const float ha = (float)sH2[M0 * HSTR + nB + nl];
      const float hb = (float)sH2[M0 * HSTR + nB + 16 + nl];
      const float hc = (float)sH2[M1 * HSTR + nB + nl];
      const float hd = (float)sH2[M1 * HSTR + nB + 16 + nl];
      sDZ[M0 * HSTR + nB + nl]      = (half_t)(c00[v] * (1.0f - ha * ha));
      sDZ[M0 * HSTR + nB + 16 + nl] = (half_t)(c01[v] * (1.0f - hb * hb));
      sDZ[M1 * HSTR + nB + nl]      = (half_t)(c10[v] * (1.0f - hc * hc));
      sDZ[M1 * HSTR + nB + 16 + nl] = (half_t)(c11[v] * (1.0f - hd * hd));
    }
    __syncthreads();

    // ---- backward: dh1 = dz2 @ W2 ; dz1 = dh1 * (1 - h1^2) ----
    gemm4(sDZ, launder_g(w2t), nullptr, lane, nB, c00, c01, c10, c11);
    __syncthreads();
#pragma unroll
    for (int v = 0; v < 8; ++v) {
      const int M0 = v + 8 * g;
      const int M1 = 16 + M0;
      const float ha = (float)sH1[M0 * HSTR + nB + nl];
      const float hb = (float)sH1[M0 * HSTR + nB + 16 + nl];
      const float hc = (float)sH1[M1 * HSTR + nB + nl];
      const float hd = (float)sH1[M1 * HSTR + nB + 16 + nl];
      sDZ[M0 * HSTR + nB + nl]      = (half_t)(c00[v] * (1.0f - ha * ha));
      sDZ[M0 * HSTR + nB + 16 + nl] = (half_t)(c01[v] * (1.0f - hb * hb));
      sDZ[M1 * HSTR + nB + nl]      = (half_t)(c10[v] * (1.0f - hc * hc));
      sDZ[M1 * HSTR + nB + 16 + nl] = (half_t)(c11[v] * (1.0f - hd * hd));
    }
    __syncthreads();

    // ---- g = dz1 @ W1  (256 -> 4 contraction, VALU) ----
    if (tid < TILEB * 4) {
      const int row = tid >> 2, j = tid & 3;
      float acc = 0.0f;
      for (int n = 0; n < HDIM; ++n)
        acc += (float)sDZ[row * HSTR + n] * W1[n * 4 + j];
      sG[row][j] = acc;
    }
    __syncthreads();

    // ---- Verlet + adaptive correction, write trajectory ----
    if (tid < TILEB) {
      const int m = tid;
      const float g0 = sG[m][0], g1v = sG[m][1], g2v = sG[m][2], g3 = sG[m][3];
      const float cr0 = g1v, cr1 = -g0, cr2 = g3, cr3 = -g2v;
      const float nrm = sqrtf(cr0*cr0 + cr1*cr1 + cr2*cr2 + cr3*cr3);
      const float adapt = dt * fminf(fmaxf(1.0f - BETA_C * nrm, 0.5f), 1.0f);
      const float q1 = sS[m][0], p1 = sS[m][1], q2 = sS[m][2], p2 = sS[m][3];
      const float F1 = -q1 * (1.0f + 2.0f * q2);
      const float F2 = -(q2 + q1 * q1 - q2 * q2);
      const float p1h = p1 + 0.5f * dt * F1;
      const float p2h = p2 + 0.5f * dt * F2;
      const float q1n = q1 + dt * p1h;
      const float q2n = q2 + dt * p2h;
      const float F1n = -q1n * (1.0f + 2.0f * q2n);
      const float F2n = -(q2n + q1n * q1n - q2n * q2n);
      const float p1n = p1h + 0.5f * dt * F1n;
      const float p2n = p2h + 0.5f * dt * F2n;
      const float sa = adapt * scl;
      const float ns0 = q1n + sa * cr0;
      const float ns1 = p1n + sa * cr1;
      const float ns2 = q2n + sa * cr2;
      const float ns3 = p2n + sa * cr3;
      sS[m][0] = ns0; sS[m][1] = ns1; sS[m][2] = ns2; sS[m][3] = ns3;
      float* op = out + (size_t)(b0 + m) * (NSTEPS * 4) + (size_t)t * 4;
      op[0] = ns0; op[1] = ns1; op[2] = ns2; op[3] = ns3;
    }
    __syncthreads();
  }
}

extern "C" void kernel_launch(void* const* d_in, const int* in_sizes, int n_in,
                              void* d_out, int out_size, void* d_ws, size_t ws_size,
                              hipStream_t stream) {
  const float* t_eval = (const float*)d_in[0];
  const float* state0 = (const float*)d_in[1];
  const float* W1     = (const float*)d_in[2];
  const float* b1     = (const float*)d_in[3];
  const float* W2     = (const float*)d_in[4];
  const float* b2     = (const float*)d_in[5];
  const float* W3     = (const float*)d_in[6];
  const float* b3     = (const float*)d_in[7];
  const float* W4     = (const float*)d_in[8];
  // d_in[9] = b4: constant shift of H, does not affect grad -> unused
  const float* scale  = (const float*)d_in[10];
  float* out = (float*)d_out;

  // Workspace layout (f16): w2h | w2t | w3h | w3t  = 4 * 64K halfs = 512 KB
  half_t* ws  = (half_t*)d_ws;
  half_t* w2h = ws;
  half_t* w2t = ws + 65536;
  half_t* w3h = ws + 131072;
  half_t* w3t = ws + 196608;

  prep_weights<<<512, 256, 0, stream>>>(W2, W3, w2h, w2t, w3h, w3t);
  sympnet_kernel<<<32768 / TILEB, 256, 0, stream>>>(
      t_eval, state0, W1, b1, b2, b3, W4, scale,
      w2h, w2t, w3h, w3t, out);
}
